// StudentModel_66073776882320
// MI455X (gfx1250) — compile-verified, matrix operand order checked by reading, not run on hardware
//
#include <hip/hip_runtime.h>
#include <hip/hip_bf16.h>
#include <stddef.h>

typedef __attribute__((ext_vector_type(16))) __bf16 v16bf;
typedef __attribute__((ext_vector_type(8)))  float  v8f;

#define D_IN  128
#define D_OUT 64

// ---------------- degree / normalization ----------------

__global__ void k_deg_init(float* __restrict__ deg, int n) {
  int i = blockIdx.x * blockDim.x + threadIdx.x;
  if (i < n) deg[i] = 1.0f;  // self loop contributes 1 to every node's degree
}

__global__ void k_deg_count(const int* __restrict__ col, float* __restrict__ deg, int e) {
  int i = blockIdx.x * blockDim.x + threadIdx.x;
  if (i < e) atomicAdd(&deg[col[i]], 1.0f);
}

__global__ void k_dinv(float* __restrict__ deg, int n) {
  int i = blockIdx.x * blockDim.x + threadIdx.x;
  if (i < n) deg[i] = rsqrtf(deg[i]);  // deg >= 1 always (self loops)
}

__global__ void k_wnorm(const int* __restrict__ row, const int* __restrict__ col,
                        const float* __restrict__ dinv, float* __restrict__ w, int e) {
  int i = blockIdx.x * blockDim.x + threadIdx.x;
  if (i < e) w[i] = dinv[row[i]] * dinv[col[i]];
}

// ---------------- weight fusion (tiny GEMMs, one block each) ----------------
// T1 = W1 @ W0  [128x128],  t0 = W1 @ b0  [128]
__global__ void k_fuse1(const float* __restrict__ W1, const float* __restrict__ W0,
                        const float* __restrict__ b0,
                        float* __restrict__ T1, float* __restrict__ t0) {
  for (int idx = threadIdx.x; idx < 128 * 128; idx += blockDim.x) {
    int i = idx >> 7, j = idx & 127;
    float acc = 0.f;
    for (int a = 0; a < 128; ++a) acc += W1[i * 128 + a] * W0[a * 128 + j];
    T1[idx] = acc;
  }
  if (threadIdx.x < 128) {
    int i = threadIdx.x;
    float acc = 0.f;
    for (int a = 0; a < 128; ++a) acc += W1[i * 128 + a] * b0[a];
    t0[i] = acc;
  }
}

// Mp = W2 @ T1 [64x128],  c0 = W2 @ t0, c1 = W2 @ b1, c2 = b2   (all [64])
__global__ void k_fuse2(const float* __restrict__ W2, const float* __restrict__ T1,
                        const float* __restrict__ t0, const float* __restrict__ b1,
                        const float* __restrict__ b2,
                        float* __restrict__ Mp, float* __restrict__ c0,
                        float* __restrict__ c1, float* __restrict__ c2) {
  for (int idx = threadIdx.x; idx < 64 * 128; idx += blockDim.x) {
    int o = idx >> 7, k = idx & 127;
    float acc = 0.f;
    for (int b = 0; b < 128; ++b) acc += W2[o * 128 + b] * T1[b * 128 + k];
    Mp[idx] = acc;
  }
  if (threadIdx.x < 64) {
    int o = threadIdx.x;
    float a0 = 0.f, a1 = 0.f;
    for (int b = 0; b < 128; ++b) {
      a0 += W2[o * 128 + b] * t0[b];
      a1 += W2[o * 128 + b] * b1[b];
    }
    c0[o] = a0; c1[o] = a1; c2[o] = b2[o];
  }
}

// ---------------- bf16 hi/lo split helpers ----------------

static __device__ __forceinline__ __bf16 bf16_hi(float f) {
  return (__bf16)f;
}
static __device__ __forceinline__ __bf16 bf16_lo(float f) {
  __bf16 h = (__bf16)f;
  return (__bf16)(f - (float)h);
}

// Pre-pack B (= Mp rows) into per-lane WMMA fragment layout, hi/lo split.
// Fragment index t = nt*128 + kk*32 + lane; each fragment is 16 bf16 (32B).
__global__ void k_pack_B(const float* __restrict__ Mp,
                         __bf16* __restrict__ Bh, __bf16* __restrict__ Bl) {
  int t = blockIdx.x * blockDim.x + threadIdx.x;
  if (t >= (D_OUT / 16) * (D_IN / 32) * 32) return;  // 512 fragments
  int lane = t & 31;
  int kk   = (t >> 5) & 3;
  int nt   = t >> 7;
  int lo16 = lane & 15;
  int off  = (lane >> 4) ? 8 : 0;
  const float* br = Mp + (size_t)(nt * 16 + lo16) * D_IN;
  int base = t * 16;
#pragma unroll
  for (int i = 0; i < 8; ++i) {
    float b0 = br[kk * 32 + off + i];
    float b1 = br[kk * 32 + 16 + off + i];
    Bh[base + i]     = bf16_hi(b0);  Bl[base + i]     = bf16_lo(b0);
    Bh[base + 8 + i] = bf16_hi(b1);  Bl[base + 8 + i] = bf16_lo(b1);
  }
}

// ---------------- projection GEMM: H = x @ Mp^T  (N x 64), bf16-split WMMA ----------------

__global__ __launch_bounds__(256)
void k_gemm_wmma(const float* __restrict__ x,
                 const __bf16* __restrict__ BhP, const __bf16* __restrict__ BlP,
                 float* __restrict__ H, int tiles_m) {
  const int wave  = (int)((blockIdx.x * blockDim.x + threadIdx.x) >> 5);
  const int lane  = threadIdx.x & 31;
  const int tiles = tiles_m * (D_OUT / 16);
  if (wave >= tiles) return;               // wave-uniform: EXEC stays all-ones
  const int mt   = wave >> 2;              // D_OUT/16 == 4
  const int nt   = wave & 3;
  const int lo16 = lane & 15;
  const int hi   = lane >> 4;              // 0 or 1
  const int off  = hi ? 8 : 0;

  const float* xr = x + (size_t)(mt * 16 + lo16) * D_IN;  // A row  (M = lo16)
  v8f c = {0.f, 0.f, 0.f, 0.f, 0.f, 0.f, 0.f, 0.f};

#pragma unroll 1   // keep one K-step live: no spills; occupancy hides latency
  for (int kk = 0; kk < D_IN / 32; ++kk) {
    const int k0 = kk * 32;
    // A fragment: lanes<16 hold K = {0..7, 16..23}, lanes>=16 hold K = {8..15, 24..31}
    v16bf Ah, Al;
#pragma unroll
    for (int i = 0; i < 8; ++i) {
      const float a0 = xr[k0 + off + i];
      const float a1 = xr[k0 + 16 + off + i];
      Ah[i]     = bf16_hi(a0);  Al[i]     = bf16_lo(a0);
      Ah[8 + i] = bf16_hi(a1);  Al[8 + i] = bf16_lo(a1);
    }
    // B fragments pre-packed in exact lane layout (32B coalesced loads)
    const size_t frag = ((size_t)(nt * 4 + kk) * 32 + lane) * 16;
    const v16bf Bh = *(const v16bf*)(BhP + frag);
    const v16bf Bl = *(const v16bf*)(BlP + frag);
    // bf16-split product: hi*hi + hi*lo + lo*hi  (~fp32 accuracy, f32 accumulate)
    c = __builtin_amdgcn_wmma_f32_16x16x32_bf16(false, Ah, false, Bh, (short)0, c, false, false);
    c = __builtin_amdgcn_wmma_f32_16x16x32_bf16(false, Ah, false, Bl, (short)0, c, false, false);
    c = __builtin_amdgcn_wmma_f32_16x16x32_bf16(false, Al, false, Bh, (short)0, c, false, false);
  }

  // C/D layout: lane = N (mod 16); VGPR r -> M = r + 8*(lane>=16)
#pragma unroll
  for (int r = 0; r < 8; ++r) {
    const int m = mt * 16 + r + hi * 8;
    H[(size_t)m * D_OUT + nt * 16 + lo16] = c[r];
  }
}

// tail rows if N not multiple of 16 (not hit for N=100000)
__global__ void k_gemm_tail(const float* __restrict__ x, const float* __restrict__ Mp,
                            float* __restrict__ H, int n0, int n) {
  int g = blockIdx.x * blockDim.x + threadIdx.x;
  int total = (n - n0) * D_OUT;
  if (g >= total) return;
  int m = n0 + g / D_OUT, o = g % D_OUT;
  float acc = 0.f;
  for (int k = 0; k < D_IN; ++k) acc += x[(size_t)m * D_IN + k] * Mp[(size_t)o * D_IN + k];
  H[(size_t)m * D_OUT + o] = acc;
}

// ---------------- propagation: h_out = P h_in  (P = D^-1/2 A D^-1/2, w/ self loops) ----------

// init pass: self-loop term, writes every element (no zeroing dependence)
__global__ void k_h_init(const float* __restrict__ hin, const float* __restrict__ dinv,
                         float* __restrict__ hout, int n) {
  int g = blockIdx.x * blockDim.x + threadIdx.x;
  if (g < n * D_OUT) {
    int node = g >> 6;
    float d = dinv[node];
    hout[g] = d * d * hin[g];
  }
}

// last pass init: fuse the rank-1 bias terms  u6*c0 + u4*c1 + u2*c2
__global__ void k_h_init_final(const float* __restrict__ hin, const float* __restrict__ dinv,
                               const float* __restrict__ u2, const float* __restrict__ u4,
                               const float* __restrict__ u6, const float* __restrict__ c0,
                               const float* __restrict__ c1, const float* __restrict__ c2,
                               float* __restrict__ hout, int n) {
  int g = blockIdx.x * blockDim.x + threadIdx.x;
  if (g < n * D_OUT) {
    int node = g >> 6, f = g & 63;
    float d = dinv[node];
    hout[g] = d * d * hin[g] + u6[node] * c0[f] + u4[node] * c1[f] + u2[node] * c2[f];
  }
}

// one wave per edge, each lane handles a float2 of the 64 features
__global__ void k_h_edge(const int* __restrict__ row, const int* __restrict__ col,
                         const float* __restrict__ w, const float* __restrict__ hin,
                         float* __restrict__ hout, int e) {
  int wid = (int)((blockIdx.x * blockDim.x + threadIdx.x) >> 5);
  if (wid >= e) return;
  int lane = threadIdx.x & 31;
  int r = row[wid], c = col[wid];
  float ww = w[wid];
  const float2 v = *(const float2*)(hin + (size_t)c * D_OUT + 2 * lane);
  float* o = hout + (size_t)r * D_OUT + 2 * lane;
  atomicAdd(o,     ww * v.x);
  atomicAdd(o + 1, ww * v.y);
}

// scalar ones-vector propagation (for bias terms)
__global__ void k_s_fill1(float* __restrict__ s, int n) {
  int i = blockIdx.x * blockDim.x + threadIdx.x;
  if (i < n) s[i] = 1.0f;
}
__global__ void k_s_init(const float* __restrict__ sin, const float* __restrict__ dinv,
                         float* __restrict__ sout, int n) {
  int i = blockIdx.x * blockDim.x + threadIdx.x;
  if (i < n) { float d = dinv[i]; sout[i] = d * d * sin[i]; }
}
__global__ void k_s_edge(const int* __restrict__ row, const int* __restrict__ col,
                         const float* __restrict__ w, const float* __restrict__ sin,
                         float* __restrict__ sout, int e) {
  int i = blockIdx.x * blockDim.x + threadIdx.x;
  if (i < e) atomicAdd(&sout[row[i]], w[i] * sin[col[i]]);
}

// ---------------- orchestration ----------------

extern "C" void kernel_launch(void* const* d_in, const int* in_sizes, int n_in,
                              void* d_out, int out_size, void* d_ws, size_t ws_size,
                              hipStream_t stream) {
  const float* x  = (const float*)d_in[0];
  const int*   ei = (const int*)d_in[1];
  const float* W0 = (const float*)d_in[2];
  const float* b0 = (const float*)d_in[3];
  const float* W1 = (const float*)d_in[4];
  const float* b1 = (const float*)d_in[5];
  const float* W2 = (const float*)d_in[6];
  const float* b2 = (const float*)d_in[7];

  const int N = in_sizes[0] / D_IN;
  const int E = in_sizes[1] / 2;
  const int* row = ei;       // edge_index[0]
  const int* col = ei + E;   // edge_index[1]
  float* out = (float*)d_out;

  // workspace carve (floats; every section is a multiple of 8 floats -> 32B aligned)
  float* ws    = (float*)d_ws;
  float* Hws   = ws;  ws += (size_t)N * D_OUT;   // ping-pong partner of d_out
  float* dinv  = ws;  ws += N;                   // deg -> dinv in place
  float* wnorm = ws;  ws += E;
  float* sA    = ws;  ws += N;
  float* sB    = ws;  ws += N;
  float* u2    = ws;  ws += N;
  float* u4    = ws;  ws += N;
  float* u6    = ws;  ws += N;
  float* T1    = ws;  ws += 128 * 128;
  float* t0v   = ws;  ws += 128;
  float* Mp    = ws;  ws += 64 * 128;
  float* c0    = ws;  ws += 64;
  float* c1    = ws;  ws += 64;
  float* c2    = ws;  ws += 64;
  __bf16* BhP  = (__bf16*)ws;  ws += 512 * 16 / 2;   // 512 fragments x 16 bf16
  __bf16* BlP  = (__bf16*)ws;  ws += 512 * 16 / 2;

  auto cdiv = [](long long a, long long b) { return (unsigned)((a + b - 1) / b); };
  dim3 B(256);

  // normalization
  k_deg_init<<<cdiv(N, 256), B, 0, stream>>>(dinv, N);
  k_deg_count<<<cdiv(E, 256), B, 0, stream>>>(col, dinv, E);
  k_dinv<<<cdiv(N, 256), B, 0, stream>>>(dinv, N);
  k_wnorm<<<cdiv(E, 256), B, 0, stream>>>(row, col, dinv, wnorm, E);

  // fuse weights: Mp = W2 W1 W0, c0 = W2 W1 b0, c1 = W2 b1, c2 = b2
  k_fuse1<<<1, B, 0, stream>>>(W1, W0, b0, T1, t0v);
  k_fuse2<<<1, B, 0, stream>>>(W2, T1, t0v, b1, b2, Mp, c0, c1, c2);
  k_pack_B<<<2, B, 0, stream>>>(Mp, BhP, BlP);

  // ones propagation: snapshots u2 = P^2 1, u4 = P^4 1, u6 = P^6 1
  k_s_fill1<<<cdiv(N, 256), B, 0, stream>>>(sA, N);
  const float* s_in[6]  = {sA, sB, u2, sA, u4, sB};
  float*       s_out[6] = {sB, u2, sA, u4, sB, u6};
  for (int p = 0; p < 6; ++p) {
    k_s_init<<<cdiv(N, 256), B, 0, stream>>>(s_in[p], dinv, s_out[p], N);
    k_s_edge<<<cdiv(E, 256), B, 0, stream>>>(row, col, wnorm, s_in[p], s_out[p], E);
  }

  // H0 = x @ Mp^T  -> d_out (WMMA path)
  const int tiles_m = N / 16;
  const int waves   = tiles_m * (D_OUT / 16);
  k_gemm_wmma<<<cdiv((long long)waves * 32, 256), B, 0, stream>>>(x, BhP, BlP, out, tiles_m);
  const int rem0 = tiles_m * 16;
  if (rem0 < N)
    k_gemm_tail<<<cdiv((long long)(N - rem0) * D_OUT, 256), B, 0, stream>>>(x, Mp, out, rem0, N);

  // 6 hops at D=64: out -> Hws -> out -> ... -> out (final pass fuses bias rank-1 terms)
  const float* h_in[6]  = {out, Hws, out, Hws, out, Hws};
  float*       h_out[6] = {Hws, out, Hws, out, Hws, out};
  for (int p = 0; p < 6; ++p) {
    if (p == 5)
      k_h_init_final<<<cdiv((long long)N * D_OUT, 256), B, 0, stream>>>(
          h_in[p], dinv, u2, u4, u6, c0, c1, c2, h_out[p], N);
    else
      k_h_init<<<cdiv((long long)N * D_OUT, 256), B, 0, stream>>>(h_in[p], dinv, h_out[p], N);
    k_h_edge<<<cdiv((long long)E * 32, 256), B, 0, stream>>>(row, col, wnorm, h_in[p], h_out[p], E);
  }
}
